// AnalogyBasedEstimation_50002009260089
// MI455X (gfx1250) — matrix operand order repaired and checked by reading, not compiled, not run on hardware
//
#include <hip/hip_runtime.h>

// ---------------------------------------------------------------------------
// KNN (AnalogyBasedEstimation): dist = sqrt(|x|^2 + |w*T|^2) - 2 x@T^T,
// top-3 smallest per row, reshape-quirk label vote, one-hot.
// GEMM: v_wmma_f32_16x16x32_bf16; B-tiles staged through LDS with
// global_load_async_to_lds_b128 TRIPLE-buffering (s_wait_asynccnt fences),
// B fragments preloaded to registers so ds latency hides under the WMMAs.
// ---------------------------------------------------------------------------

typedef __attribute__((ext_vector_type(16))) __bf16 v16bf;
typedef __attribute__((ext_vector_type(8)))  float  v8f;
typedef unsigned short ushort_t;

#define B_DIM 2048
#define N_DIM 65536
#define F_DIM 256
#define NLAB  100
#define NSPLIT 8          // N-dimension split across blocks (train is L2-resident)
#define WAVES 4           // waves per GEMM block
#define NBUF  3           // async pipeline depth (2 tiles in flight)
#define FLT_BIG 3.402823466e38f

union Frag16 { v16bf bf; uint4 q[2]; };

__device__ __forceinline__ ushort_t f2bf(float f) {
  unsigned u = __float_as_uint(f);
  unsigned r = u + 0x7FFFu + ((u >> 16) & 1u);   // round-to-nearest-even
  return (ushort_t)(r >> 16);
}
__device__ __forceinline__ unsigned pack2(float a, float b) {
  return (unsigned)f2bf(a) | ((unsigned)f2bf(b) << 16);
}

// Issue one 8KB contiguous tile (16 train rows x 512B) as 16 async b128
// copies: the same imm offset is applied to both global and LDS addresses.
template <int I> struct AsyncTile {
  static __device__ __forceinline__ void go(unsigned lds, unsigned long long ga) {
    AsyncTile<I - 1>::go(lds, ga);
    asm volatile("global_load_async_to_lds_b128 %0, %1, off offset:%2"
                 :: "v"(lds), "v"(ga), "n"((I - 1) * 512) : "memory");
  }
};
template <> struct AsyncTile<0> {
  static __device__ __forceinline__ void go(unsigned, unsigned long long) {}
};

__device__ __forceinline__ void issue_tile_async(const ushort_t* gsrc,
                                                 const ushort_t* lds_base,
                                                 int lane) {
  unsigned lds = (unsigned)(size_t)lds_base + (unsigned)(lane * 16);
  unsigned long long ga = (unsigned long long)(size_t)gsrc
                        + (unsigned long long)(lane * 16);
  AsyncTile<16>::go(lds, ga);
}

// One wave per row: squared-norm (optionally feature-weighted) + bf16 convert.
__global__ __launch_bounds__(256) void prep_rows(
    const float* __restrict__ src, const float* __restrict__ feat,
    ushort_t* __restrict__ dstbf, float* __restrict__ norms,
    int nrows, int use_feat) {
  int wave = threadIdx.x >> 5;
  int lane = threadIdx.x & 31;
  int row  = blockIdx.x * 8 + wave;
  if (row >= nrows) return;

  const float* r = src + (size_t)row * F_DIM + lane * 8;
  float4 a = *(const float4*)(r);
  float4 b = *(const float4*)(r + 4);

  uint4 p;
  p.x = pack2(a.x, a.y); p.y = pack2(a.z, a.w);
  p.z = pack2(b.x, b.y); p.w = pack2(b.z, b.w);
  *(uint4*)(dstbf + (size_t)row * F_DIM + lane * 8) = p;

  float s;
  if (use_feat) {
    const float* fp = feat + lane * 8;
    float4 fa = *(const float4*)(fp);
    float4 fb = *(const float4*)(fp + 4);
    float w0 = fa.x * a.x, w1 = fa.y * a.y, w2 = fa.z * a.z, w3 = fa.w * a.w;
    float w4 = fb.x * b.x, w5 = fb.y * b.y, w6 = fb.z * b.z, w7 = fb.w * b.w;
    s = w0*w0 + w1*w1 + w2*w2 + w3*w3 + w4*w4 + w5*w5 + w6*w6 + w7*w7;
  } else {
    s = a.x*a.x + a.y*a.y + a.z*a.z + a.w*a.w +
        b.x*b.x + b.y*b.y + b.z*b.z + b.w*b.w;
  }
  #pragma unroll
  for (int o = 16; o > 0; o >>= 1) s += __shfl_xor(s, o, 32);
  if (lane == 0) norms[row] = s;
}

// Fused bf16-WMMA GEMM + per-row top-3 over one N-split.
// grid = (NSPLIT, B/16); 4 waves stripe this split's 512 column tiles.
__global__ __launch_bounds__(128, 1) void knn_gemm_top3(
    const ushort_t* __restrict__ xbf, const ushort_t* __restrict__ tbf,
    const float* __restrict__ leftN, const float* __restrict__ rightN,
    float* __restrict__ candv, int* __restrict__ candi) {
  __shared__ __align__(16) ushort_t bbuf[WAVES][NBUF][16 * F_DIM];  // 96 KB
  __shared__ float cv[16][WAVES][16][3];
  __shared__ int   ci[16][WAVES][16][3];

  const int tid  = threadIdx.x;
  const int wave = tid >> 5;
  const int lane = tid & 31;
  const int half = lane >> 4;
  const int l15  = lane & 15;
  const int split = blockIdx.x;
  const int mbase = blockIdx.y * 16;
  const int tbase = split * (N_DIM / 16 / NSPLIT);    // 512 tiles per split
  const int NITER = (N_DIM / 16 / NSPLIT) / WAVES;    // 128 tiles per wave

  // A fragments: 16x256 x-tile, register-resident for the whole kernel.
  // 16-bit A layout: lane L<16 holds M=L, K={k0..k0+7, k0+16..k0+23};
  // lane L+16 holds M=L, K={k0+8..k0+15, k0+24..k0+31}.
  Frag16 afr[8];
  const ushort_t* xrow = xbf + (size_t)(mbase + l15) * F_DIM + half * 8;
  #pragma unroll
  for (int k = 0; k < 8; ++k) {
    afr[k].q[0] = *(const uint4*)(xrow + k * 32);
    afr[k].q[1] = *(const uint4*)(xrow + k * 32 + 16);
  }

  float lf[8];
  #pragma unroll
  for (int v = 0; v < 8; ++v) lf[v] = leftN[mbase + v + 8 * half];

  float bv0[8], bv1[8], bv2[8];
  int   bi0[8], bi1[8], bi2[8];
  #pragma unroll
  for (int v = 0; v < 8; ++v) {
    bv0[v] = bv1[v] = bv2[v] = FLT_BIG;
    bi0[v] = bi1[v] = bi2[v] = 0;
  }

  // Prime the async pipeline two tiles deep (32 outstanding async ops max).
  issue_tile_async(tbf + (size_t)(tbase + wave) * 16 * F_DIM,
                   &bbuf[wave][0][0], lane);
  if (NITER > 1)
    issue_tile_async(tbf + (size_t)(tbase + wave + WAVES) * 16 * F_DIM,
                     &bbuf[wave][1][0], lane);

  int cur = 0, fill = 2;
  for (int i = 0; i < NITER; ++i) {
    const int t    = tbase + wave + WAVES * i;
    const int ncol = t * 16 + l15;
    const float rt = rightN[ncol];

    // Keep two tiles in flight; fence so the current tile is complete.
    if (i + 2 < NITER) {
      issue_tile_async(tbf + (size_t)(t + 2 * WAVES) * 16 * F_DIM,
                       &bbuf[wave][fill][0], lane);
      asm volatile("s_wait_asynccnt 32" ::: "memory");
    } else if (i + 1 < NITER) {
      asm volatile("s_wait_asynccnt 16" ::: "memory");
    } else {
      asm volatile("s_wait_asynccnt 0" ::: "memory");
    }

    // Hardware sqrt (v_sqrt_f32, TRANS pipe) overlaps with the matrix pipe.
    float srt[8];
    #pragma unroll
    for (int v = 0; v < 8; ++v) srt[v] = __builtin_amdgcn_sqrtf(lf[v] + rt);

    // Preload ALL 8 B fragments (16 clause-able ds_load_b128) so partial
    // dscnt waits hide LDS latency under the WMMA chain.
    // 16-bit B (KxN) layout: lane L holds N=l15, K = {half*8..half*8+7, 16+half*8..16+half*8+7}.
    const ushort_t* trow = &bbuf[wave][cur][l15 * F_DIM + half * 8];
    Frag16 bfr[8];
    #pragma unroll
    for (int k = 0; k < 8; ++k) {
      bfr[k].q[0] = *(const uint4*)(trow + k * 32);
      bfr[k].q[1] = *(const uint4*)(trow + k * 32 + 16);
    }

    v8f c = {};
    #pragma unroll
    for (int k = 0; k < 8; ++k)
      c = __builtin_amdgcn_wmma_f32_16x16x32_bf16(
              false, afr[k].bf, false, bfr[k].bf, (short)0, c, false, false);

    #pragma unroll
    for (int v = 0; v < 8; ++v) {
      float d = srt[v] - 2.0f * c[v];
      if (d < bv2[v]) {                       // rare after warm-up: 1 cmpx
        bool lt0 = d < bv0[v];
        bool lt1 = d < bv1[v];
        float n2v = lt1 ? bv1[v] : d;   int n2i = lt1 ? bi1[v] : ncol;
        float n1v = lt0 ? bv0[v] : (lt1 ? d : bv1[v]);
        int   n1i = lt0 ? bi0[v] : (lt1 ? ncol : bi1[v]);
        float n0v = lt0 ? d : bv0[v];   int n0i = lt0 ? ncol : bi0[v];
        bv0[v] = n0v; bi0[v] = n0i;
        bv1[v] = n1v; bi1[v] = n1i;
        bv2[v] = n2v; bi2[v] = n2i;
      }
    }
    cur  = (cur == NBUF - 1) ? 0 : cur + 1;
    fill = (fill == NBUF - 1) ? 0 : fill + 1;
  }

  // Dump per-lane candidates. C layout: VGPR v -> row (v + 8*half), col l15.
  #pragma unroll
  for (int v = 0; v < 8; ++v) {
    int m = v + 8 * half;
    cv[m][wave][l15][0] = bv0[v]; ci[m][wave][l15][0] = bi0[v];
    cv[m][wave][l15][1] = bv1[v]; ci[m][wave][l15][1] = bi1[v];
    cv[m][wave][l15][2] = bv2[v]; ci[m][wave][l15][2] = bi2[v];
  }
  __syncthreads();

  if (tid < 16) {
    const int m = tid;
    float v0 = FLT_BIG, v1 = FLT_BIG, v2 = FLT_BIG;
    int   i0 = 0, i1 = 0, i2 = 0;
    for (int w = 0; w < WAVES; ++w)
      for (int c = 0; c < 16; ++c)
        for (int j = 0; j < 3; ++j) {
          float d = cv[m][w][c][j];
          int   i = ci[m][w][c][j];
          if (d < v0)      { v2=v1; i2=i1; v1=v0; i1=i0; v0=d; i0=i; }
          else if (d < v1) { v2=v1; i2=i1; v1=d;  i1=i; }
          else if (d < v2) { v2=d;  i2=i; }
        }
    size_t o = ((size_t)(mbase + m) * NSPLIT + split) * 3;
    candv[o] = v0; candv[o + 1] = v1; candv[o + 2] = v2;
    candi[o] = i0; candi[o + 1] = i1; candi[o + 2] = i2;
  }
}

// Merge the NSPLIT partial top-3 lists per row into the final ordered top-3.
__global__ __launch_bounds__(256) void merge_topk(
    const float* __restrict__ candv, const int* __restrict__ candi,
    int* __restrict__ topk) {
  int b = blockIdx.x * blockDim.x + threadIdx.x;
  if (b >= B_DIM) return;
  float v0 = FLT_BIG, v1 = FLT_BIG, v2 = FLT_BIG;
  int   i0 = 0, i1 = 0, i2 = 0;
  #pragma unroll
  for (int s = 0; s < NSPLIT; ++s)
    #pragma unroll
    for (int j = 0; j < 3; ++j) {
      float d = candv[((size_t)b * NSPLIT + s) * 3 + j];
      int   i = candi[((size_t)b * NSPLIT + s) * 3 + j];
      if (d < v0)      { v2=v1; i2=i1; v1=v0; i1=i0; v0=d; i0=i; }
      else if (d < v1) { v2=v1; i2=i1; v1=d;  i1=i; }
      else if (d < v2) { v2=d;  i2=i; }
    }
  int* o = topk + (size_t)b * 3;
  o[0] = i0; o[1] = i1; o[2] = i2;     // ascending distance
}

// Faithful tail: labels[B,3] row-major flat -> reshape [3,B]; sum//3; one-hot.
__global__ __launch_bounds__(256) void labels_onehot(
    const long long* __restrict__ labels,
    const int* __restrict__ topk_idx,
    float* __restrict__ out) {
  int b = blockIdx.x * blockDim.x + threadIdx.x;
  if (b >= B_DIM) return;
  int s = 0;
  #pragma unroll
  for (int i = 0; i < 3; ++i) {
    int p = i * B_DIM + b;          // flat position in [B,3]
    int r = p / 3, c = p % 3;
    s += (int)labels[topk_idx[r * 3 + c]];
  }
  int lab = s / 3;                  // integer mean (labels >= 0)
  float* o = out + (size_t)b * NLAB;
  for (int j = 0; j < NLAB; ++j) o[j] = (j == lab) ? 1.0f : 0.0f;
}

extern "C" void kernel_launch(void* const* d_in, const int* in_sizes, int n_in,
                              void* d_out, int out_size, void* d_ws, size_t ws_size,
                              hipStream_t stream) {
  const float*     x    = (const float*)d_in[0];
  const float*     tr   = (const float*)d_in[1];
  const float*     feat = (const float*)d_in[2];
  const long long* lbl  = (const long long*)d_in[3];
  (void)in_sizes; (void)n_in; (void)out_size; (void)ws_size;

  char* ws = (char*)d_ws;
  ushort_t* tbf   = (ushort_t*)(ws);                       // 33,554,432 B
  ushort_t* xbf   = (ushort_t*)(ws + 33554432);            //  1,048,576 B
  float*    right = (float*)   (ws + 34603008);            //    262,144 B
  float*    left  = (float*)   (ws + 34865152);            //      8,192 B
  float*    candv = (float*)   (ws + 34873344);            //    196,608 B
  int*      candi = (int*)     (ws + 35069952);            //    196,608 B
  int*      topk  = (int*)     (ws + 35266560);            //     24,576 B

  prep_rows<<<N_DIM / 8, 256, 0, stream>>>(tr, feat, tbf, right, N_DIM, 1);
  prep_rows<<<B_DIM / 8, 256, 0, stream>>>(x,  feat, xbf, left,  B_DIM, 0);
  dim3 grid(NSPLIT, B_DIM / 16);
  knn_gemm_top3<<<grid, 32 * WAVES, 0, stream>>>(xbf, tbf, left, right,
                                                 candv, candi);
  merge_topk<<<B_DIM / 256, 256, 0, stream>>>(candv, candi, topk);
  labels_onehot<<<B_DIM / 256, 256, 0, stream>>>(lbl, topk, (float*)d_out);
}
